// HybridModel_45569603011098
// MI455X (gfx1250) — matrix operand order repaired and checked by reading, not compiled, not run on hardware
//
#include <hip/hip_runtime.h>
#include <hip/hip_bf16.h>
#include <stdint.h>

typedef __attribute__((ext_vector_type(16))) _Float16     v16h;
typedef __attribute__((ext_vector_type(8)))  float        v8f;
typedef __attribute__((ext_vector_type(4)))  unsigned int u32x4;
typedef __attribute__((ext_vector_type(8)))  int          i32x8;
typedef __attribute__((ext_vector_type(4)))  int          i32x4;

typedef __attribute__((address_space(1))) int  glb_i32;
typedef __attribute__((address_space(3))) int  lds_i32;
typedef __attribute__((address_space(3))) float lds_f32;

#define HAVE_ASYNC_LDS __has_builtin(__builtin_amdgcn_global_load_async_to_lds_b32)
#define HAVE_TDM       __has_builtin(__builtin_amdgcn_tensor_load_to_lds)

// ---------------------------------------------------------------------------
// counter waits (guarded; fall back to inline asm mnemonics from the ISA)
// ---------------------------------------------------------------------------
__device__ __forceinline__ void wait_async0()
{
#if __has_builtin(__builtin_amdgcn_s_wait_asynccnt)
    __builtin_amdgcn_s_wait_asynccnt(0);
#elif HAVE_ASYNC_LDS
    asm volatile("s_wait_asynccnt 0x0" ::: "memory");
#endif
}
__device__ __forceinline__ void wait_tensor0()
{
#if __has_builtin(__builtin_amdgcn_s_wait_tensorcnt)
    __builtin_amdgcn_s_wait_tensorcnt(0);
#elif HAVE_TDM
    asm volatile("s_wait_tensorcnt 0x0" ::: "memory");
#endif
}

// ---------------------------------------------------------------------------
// one f32 element: global -> LDS (async direct-to-LDS when available);
// invalid lanes write zero to their (disjoint) LDS slot instead.
// ---------------------------------------------------------------------------
__device__ __forceinline__ void stage_elem(float* lds, const float* g, bool valid)
{
#if HAVE_ASYNC_LDS
    if (valid)
        __builtin_amdgcn_global_load_async_to_lds_b32(
            (glb_i32*)(float*)g, (lds_i32*)lds, 0, 0);
    else
        *lds = 0.f;
#else
    *lds = valid ? *g : 0.f;
#endif
}

// ---------------------------------------------------------------------------
// TDM weight-tile load: 16 OC rows x 32 channels of OIHW weights for one
// (ky,kx) tap.  3D tile: x=1 elem, y=32 channels (stride 9 elems),
// z=16 OC rows (stride C*9 elems); OOB rows/channels zero-filled by TDM.
// ---------------------------------------------------------------------------
#if HAVE_TDM
__device__ __forceinline__ void tdm_load_A(float* ldsA, const float* gA,
                                           int remC, int remOC, int rowStrideElem)
{
    unsigned long long ga = (unsigned long long)(uintptr_t)gA;
    unsigned lds_off = (unsigned)(uintptr_t)(lds_f32*)ldsA;

    u32x4 g0;
    g0[0] = 1u;                                        // count=1 (valid D#)
    g0[1] = lds_off;                                   // lds_addr
    g0[2] = (unsigned)(ga & 0xffffffffu);              // global_addr lo
    g0[3] = (unsigned)((ga >> 32) & 0x01ffffffu) | (2u << 30);  // hi | type=2

    i32x8 g1;
    g1[0] = 2 << 16;                                   // data_size = 4B
    g1[1] = 1 << 16;                                   // tensor_dim0 = 1
    g1[2] = (remC & 0xffff) << 16;                     // tensor_dim1 lo16
    g1[3] = ((remC >> 16) & 0xffff) | (1 << 16);       // dim1 hi | tile_dim0=1
    g1[4] = 32 | (16 << 16);                           // tile_dim1=32, tile_dim2=16
    g1[5] = 9;                                         // tensor_dim0_stride = 9 elems
    g1[6] = (rowStrideElem & 0xffff) << 16;            // tensor_dim1_stride lo16
    g1[7] = rowStrideElem >> 16;                       //   ... hi32

    i32x4 g2; g2[0] = remOC; g2[1] = 0; g2[2] = 0; g2[3] = 0;  // tensor_dim2
    i32x4 g3 = {};

#if __clang_major__ >= 23
    i32x8 gz = {};
    __builtin_amdgcn_tensor_load_to_lds(g0, g1, g2, g3, gz, 0);
#else
    __builtin_amdgcn_tensor_load_to_lds(g0, g1, g2, g3, 0);
#endif
}
#endif

// ---------------------------------------------------------------------------
// Implicit-GEMM conv via CDNA5 WMMA.
//   block = 256 threads = 8 waves; tile = 16 OC x 128 pixels
//   A (16x32 weight tile) staged once per block (TDM), shared by all waves
//   B (32x16 im2col tile) staged per wave via async global->LDS
//   K reduced in chunks of 32 with v_wmma_f32_16x16x32_f16 (f32 accum)
//   concat partner + nearest-2x upsample folded into input addressing
// ---------------------------------------------------------------------------
template<int KS, int STRIDE>
__global__ __launch_bounds__(256)
void conv2d_wmma(const float* __restrict__ in1, int C1, int ls1, long long n1s,
                 const float* __restrict__ in2, int C2, long long n2s,
                 const float* __restrict__ wt, const float* __restrict__ bias,
                 float* __restrict__ out,
                 int IH, int IW, int OH, int OW,
                 int pad, int OC, int act)
{
    __shared__ float As[16 * 32];       // shared weight tile (f32)
    __shared__ float Bs[8][32 * 16];    // per-wave im2col tile (f32)

    const int tid  = threadIdx.x;
    const int wave = tid >> 5;
    const int lane = tid & 31;
    const int hi   = lane >> 4;
    const int m    = lane & 15;

    const int n    = blockIdx.z;
    const int ocb  = blockIdx.y * 16;
    const int pixb = blockIdx.x * 128 + wave * 16;

    const int C   = C1 + C2;
    const int OHW = OH * OW;
    const int IH1 = IH >> ls1, IW1 = IW >> ls1;
    const long long ch1 = (long long)IH1 * IW1;
    const long long ch2 = (long long)IH * IW;
    const float* in1n = in1 + (long long)n * n1s;
    const float* in2n = in2 ? (in2 + (long long)n * n2s) : in1;
    const int cblocks = (C + 31) >> 5;
    const int wKK = KS * KS;

    // fixed per-lane B column (output pixel)
    const int  pix = pixb + m;
    const bool pv  = pix < OHW;
    const int  oh  = pv ? pix / OW : 0;
    const int  ow  = pv ? pix % OW : 0;

    float* Bw = &Bs[wave][0];
    v8f acc = {};

    #pragma unroll
    for (int ky = 0; ky < KS; ++ky) {
        #pragma unroll
        for (int kx = 0; kx < KS; ++kx) {
            const int  ih = oh * STRIDE + ky - pad;
            const int  iw = ow * STRIDE + kx - pad;
            const bool iv = pv && ih >= 0 && ih < IH && iw >= 0 && iw < IW;
            const long long sp1 = (long long)(ih >> ls1) * IW1 + (iw >> ls1);
            const long long sp2 = (long long)ih * IW + iw;

            for (int cb = 0; cb < cblocks; ++cb) {
                const int cbase = cb << 5;

                // ---- stage A: weight tile -> LDS ----
#if HAVE_TDM
                if (wave == 0)
                    tdm_load_A(As,
                               wt + (((long long)ocb * C + cbase) * KS + ky) * KS + kx,
                               C - cbase, OC - ocb, C * wKK);
#else
                #pragma unroll
                for (int i = 0; i < 2; ++i) {
                    int e = tid + i * 256;
                    int mrow = e >> 5, k = e & 31;
                    int oc = ocb + mrow, c = cbase + k;
                    stage_elem(&As[e],
                               wt + (((long long)oc * C + c) * KS + ky) * KS + kx,
                               (oc < OC) && (c < C));
                }
#endif
                // ---- stage B: im2col tile -> LDS (async direct-to-LDS) ----
                #pragma unroll
                for (int i = 0; i < 16; ++i) {
                    int k = 2 * i + hi;          // lanes 0-15: even k, 16-31: odd k
                    int c = cbase + k;
                    bool u1 = iv && (c < C1);
                    bool u2 = iv && (c >= C1) && (c < C);
                    const float* g = u1 ? (in1n + (long long)c * ch1 + sp1)
                                        : (in2n + (long long)(c - C1) * ch2 + sp2);
                    stage_elem(&Bw[k * 16 + m], g, u1 || u2);
                }

                wait_tensor0();
                wait_async0();
                __syncthreads();

                // ---- fragments (contiguous k = 16*hi + s in both A and B) + MMA ----
                v16h a, b;
                #pragma unroll
                for (int r = 0; r < 8; ++r) {
                    int ka = (hi ? 16 : 0) + 2 * r;
                    a[2 * r]     = (_Float16)As[m * 32 + ka];
                    a[2 * r + 1] = (_Float16)As[m * 32 + ka + 1];
                    int kb = (hi ? 16 : 0) + 2 * r;
                    b[2 * r]     = (_Float16)Bw[kb * 16 + m];
                    b[2 * r + 1] = (_Float16)Bw[(kb + 1) * 16 + m];
                }
                acc = __builtin_amdgcn_wmma_f32_16x16x32_f16(false, a, false, b,
                                                             (short)0, acc,
                                                             false, false);
                __syncthreads();
            }
        }
    }

    // ---- store D: VGPR r -> M = r (+8 for lanes 16-31), N = lane&15 ----
    #pragma unroll
    for (int r = 0; r < 8; ++r) {
        int oc = ocb + (hi ? 8 + r : r);
        if (oc < OC && pv) {
            float v = acc[r] + bias[oc];
            if (act == 1)      v = v > 0.f ? v : 0.f;
            else if (act == 2) v = 1.f / (1.f + __expf(-v));
            out[((long long)n * OC + oc) * OHW + pix] = v;
        }
    }
}

// ---------------------------------------------------------------------------
// 2x2 max pool, stride 2
// ---------------------------------------------------------------------------
__global__ void maxpool2_k(const float* __restrict__ in, float* __restrict__ out,
                           long long total, int C, int H, int W)
{
    long long i = (long long)blockIdx.x * blockDim.x + threadIdx.x;
    if (i >= total) return;
    int OW = W >> 1, OH = H >> 1;
    int ow = (int)(i % OW); long long r = i / OW;
    int oh = (int)(r % OH); r /= OH;
    int c  = (int)(r % C);  int n = (int)(r / C);
    const float* p = in + (((long long)(n * C + c) * H + oh * 2) * W + ow * 2);
    out[i] = fmaxf(fmaxf(p[0], p[1]), fmaxf(p[W], p[W + 1]));
}

// per-(n,c) spatial mean
__global__ void mean_hw_k(const float* __restrict__ in, float* __restrict__ out, int HW)
{
    __shared__ float s[256];
    const float* p = in + (long long)blockIdx.x * HW;
    float acc = 0.f;
    for (int i = threadIdx.x; i < HW; i += 256) acc += p[i];
    s[threadIdx.x] = acc; __syncthreads();
    for (int o = 128; o > 0; o >>= 1) {
        if (threadIdx.x < (unsigned)o) s[threadIdx.x] += s[threadIdx.x + o];
        __syncthreads();
    }
    if (threadIdx.x == 0) out[blockIdx.x] = s[0] / (float)HW;
}

// bbox_features = sigmoid(mean @ fw.T + fb)   (16x128 @ 128x4)
__global__ void fc4_sigmoid_k(const float* __restrict__ f, const float* __restrict__ w,
                              const float* __restrict__ b, float* __restrict__ out, int K)
{
    int t = threadIdx.x; if (t >= 64) return;
    int n = t >> 2, j = t & 3;
    float s = b[j];
    for (int k = 0; k < K; ++k) s += f[n * K + k] * w[j * K + k];
    out[t] = 1.f / (1.f + __expf(-s));
}

__global__ void mul_k(const float* __restrict__ a, const float* __restrict__ b,
                      float* __restrict__ o, long long n)
{
    long long i = (long long)blockIdx.x * blockDim.x + threadIdx.x;
    if (i < n) o[i] = a[i] * b[i];
}

// mask_to_bbox: one block per batch sample
__global__ void bbox_k(const float* __restrict__ mask, float* __restrict__ out)
{
    __shared__ int rmin, rmax, cmin, cmax;
    const int n = blockIdx.x, tid = threadIdx.x;
    const float* m = mask + (long long)n * 384 * 384;
    if (tid == 0) { rmin = 1 << 30; rmax = -1; cmin = 1 << 30; cmax = -1; }
    __syncthreads();
    for (int r = tid; r < 384; r += 256) {
        bool any = false;
        for (int c = 0; c < 384; ++c) if (m[r * 384 + c] > 0.5f) { any = true; break; }
        if (any) { atomicMin(&rmin, r); atomicMax(&rmax, r); }
    }
    for (int c = tid; c < 384; c += 256) {
        bool any = false;
        for (int r = 0; r < 384; ++r) if (m[r * 384 + c] > 0.5f) { any = true; break; }
        if (any) { atomicMin(&cmin, c); atomicMax(&cmax, c); }
    }
    __syncthreads();
    if (tid == 0) {
        bool valid = (rmax >= 0) && (cmax >= 0);
        float x1 = cmin / 384.f, y1 = rmin / 384.f, x2 = cmax / 384.f, y2 = rmax / 384.f;
        x1 = fminf(fmaxf(x1, 0.f), 1.f); y1 = fminf(fmaxf(y1, 0.f), 1.f);
        x2 = fminf(fmaxf(x2, 0.f), 1.f); y2 = fminf(fmaxf(y2, 0.f), 1.f);
        if (!valid) { x1 = y1 = x2 = y2 = 0.f; }
        out[n * 4 + 0] = x1; out[n * 4 + 1] = y1; out[n * 4 + 2] = x2; out[n * 4 + 3] = y2;
    }
}

// head MLP with per-batch BN (B=16); writes final_bbox to out[0:64]
__global__ __launch_bounds__(256)
void head_k(const float* __restrict__ bf, const float* __restrict__ mb,
            const float* __restrict__ w1, const float* __restrict__ b1,
            const float* __restrict__ g1, const float* __restrict__ be1,
            const float* __restrict__ w2, const float* __restrict__ b2,
            const float* __restrict__ g2, const float* __restrict__ be2,
            const float* __restrict__ w3, const float* __restrict__ b3,
            float* __restrict__ out)
{
    __shared__ float h1[16 * 256];
    __shared__ float h2[16 * 64];
    __shared__ float fbv[64];
    const int j = threadIdx.x;

    {   // layer1: (16x8)@(8x256) + relu + BN over batch
        float vals[16]; float mu = 0.f;
        for (int nn = 0; nn < 16; ++nn) {
            float s = b1[j];
            #pragma unroll
            for (int k = 0; k < 8; ++k) {
                float hv = (k < 4) ? bf[nn * 4 + k] : mb[nn * 4 + (k - 4)];
                s += hv * w1[j * 8 + k];
            }
            s = s > 0.f ? s : 0.f;
            vals[nn] = s; mu += s;
        }
        mu *= (1.f / 16.f);
        float var = 0.f;
        for (int nn = 0; nn < 16; ++nn) { float d = vals[nn] - mu; var += d * d; }
        var *= (1.f / 16.f);
        float inv = rsqrtf(var + 1e-5f);
        for (int nn = 0; nn < 16; ++nn)
            h1[nn * 256 + j] = g1[j] * (vals[nn] - mu) * inv + be1[j];
    }
    __syncthreads();
    if (j < 64) {   // layer2: (16x256)@(256x64) + relu + BN
        float vals[16]; float mu = 0.f;
        for (int nn = 0; nn < 16; ++nn) {
            float s = b2[j];
            for (int k = 0; k < 256; ++k) s += h1[nn * 256 + k] * w2[j * 256 + k];
            s = s > 0.f ? s : 0.f;
            vals[nn] = s; mu += s;
        }
        mu *= (1.f / 16.f);
        float var = 0.f;
        for (int nn = 0; nn < 16; ++nn) { float d = vals[nn] - mu; var += d * d; }
        var *= (1.f / 16.f);
        float inv = rsqrtf(var + 1e-5f);
        for (int nn = 0; nn < 16; ++nn)
            h2[nn * 64 + j] = g2[j] * (vals[nn] - mu) * inv + be2[j];
    }
    __syncthreads();
    if (j < 64) {   // layer3: (16x64)@(64x4) + sigmoid
        int nn = j >> 2, jj = j & 3;
        float s = b3[jj];
        for (int k = 0; k < 64; ++k) s += h2[nn * 64 + k] * w3[jj * 64 + k];
        fbv[j] = 1.f / (1.f + __expf(-s));
    }
    __syncthreads();
    if (j < 16) {   // monotonic bbox fix-up
        float x1 = fbv[j * 4 + 0], y1 = fbv[j * 4 + 1];
        float x2 = fbv[j * 4 + 2], y2 = fbv[j * 4 + 3];
        float dx = x2 - x1; dx = dx > 0.f ? dx : 0.f; x2 = x1 + dx + 0.001f;
        float dy = y2 - y1; dy = dy > 0.f ? dy : 0.f; y2 = y1 + dy + 0.001f;
        out[j * 4 + 0] = x1; out[j * 4 + 1] = y1;
        out[j * 4 + 2] = x2; out[j * 4 + 3] = y2;
    }
}

// ---------------------------------------------------------------------------
// host-side orchestration
// ---------------------------------------------------------------------------
static inline void conv(hipStream_t s,
                        const float* in1, int C1, int ls1, long long n1s,
                        const float* in2, int C2, long long n2s,
                        const float* wt, const float* bias, float* out,
                        int IH, int IW, int OH, int OW,
                        int KS, int stride, int pad, int OC, int act)
{
    dim3 g((OH * OW + 127) / 128, (OC + 15) / 16, 16);
    if (KS == 3 && stride == 1)
        conv2d_wmma<3, 1><<<g, 256, 0, s>>>(in1, C1, ls1, n1s, in2, C2, n2s,
                                            wt, bias, out, IH, IW, OH, OW, pad, OC, act);
    else if (KS == 3 && stride == 2)
        conv2d_wmma<3, 2><<<g, 256, 0, s>>>(in1, C1, ls1, n1s, in2, C2, n2s,
                                            wt, bias, out, IH, IW, OH, OW, pad, OC, act);
    else
        conv2d_wmma<1, 1><<<g, 256, 0, s>>>(in1, C1, ls1, n1s, in2, C2, n2s,
                                            wt, bias, out, IH, IW, OH, OW, pad, OC, act);
}

extern "C" void kernel_launch(void* const* d_in, const int* in_sizes, int n_in,
                              void* d_out, int out_size, void* d_ws, size_t ws_size,
                              hipStream_t stream)
{
    (void)in_sizes; (void)n_in; (void)out_size; (void)ws_size;
    #define IN(i) ((const float*)d_in[(i)])
    const float* x = IN(0);
    float* outp = (float*)d_out;

    const long long S384 = 384 * 384, S192 = 192 * 192, S96 = 96 * 96, S48 = 48 * 48;

    float* ws = (float*)d_ws;
    size_t off = 0;
    auto alloc = [&](long long nElem) { float* p = ws + off; off += (size_t)nElem; return p; };

    float* f1    = alloc(16 * 32 * S192);
    float* f2    = alloc(16 * 64 * S96);
    float* f3    = alloc(16 * 128 * S48);
    float* meanf = alloc(16 * 128);
    float* bboxf = alloc(64);
    float* e1    = alloc(16 * 32 * S384);
    float* p1    = alloc(16 * 32 * S192);
    float* e2    = alloc(16 * 64 * S192);
    float* p2    = alloc(16 * 64 * S96);
    float* bt    = alloc(16 * 128 * S96);
    float* d2a   = alloc(16 * 64 * S192);
    float* d1a   = alloc(16 * 32 * S384);
    float* maskb = alloc(16 * S384);
    float* atth  = alloc(16 * 16 * S384);
    float* attb  = alloc(16 * S384);
    float* refb  = alloc(16 * S384);
    float* r1    = alloc(16 * 32 * S384);
    float* r2    = alloc(16 * 16 * S384);
    float* mbbox = alloc(64);

    // ---- eff backbone: three stride-2 convs + GAP + FC(sigmoid) ----
    conv(stream, x, 3, 0, 3 * S384, nullptr, 0, 0, IN(1), IN(2), f1,
         384, 384, 192, 192, 3, 2, 0, 32, 1);
    conv(stream, f1, 32, 0, 32 * S192, nullptr, 0, 0, IN(3), IN(4), f2,
         192, 192, 96, 96, 3, 2, 0, 64, 1);
    conv(stream, f2, 64, 0, 64 * S96, nullptr, 0, 0, IN(5), IN(6), f3,
         96, 96, 48, 48, 3, 2, 0, 128, 1);
    mean_hw_k<<<16 * 128, 256, 0, stream>>>(f3, meanf, (int)S48);
    fc4_sigmoid_k<<<1, 64, 0, stream>>>(meanf, IN(7), IN(8), bboxf, 128);

    // ---- UNet on grayscale channel (x[:,0:1]) ----
    conv(stream, x, 1, 0, 3 * S384, nullptr, 0, 0, IN(9), IN(10), e1,
         384, 384, 384, 384, 3, 1, 1, 32, 1);
    {
        long long tp = 16 * 32 * S192;
        maxpool2_k<<<(unsigned)((tp + 255) / 256), 256, 0, stream>>>(e1, p1, tp, 32, 384, 384);
    }
    conv(stream, p1, 32, 0, 32 * S192, nullptr, 0, 0, IN(11), IN(12), e2,
         192, 192, 192, 192, 3, 1, 1, 64, 1);
    {
        long long tp = 16 * 64 * S96;
        maxpool2_k<<<(unsigned)((tp + 255) / 256), 256, 0, stream>>>(e2, p2, tp, 64, 192, 192);
    }
    conv(stream, p2, 64, 0, 64 * S96, nullptr, 0, 0, IN(13), IN(14), bt,
         96, 96, 96, 96, 3, 1, 1, 128, 1);
    // d2 = conv(concat(upsample2(bt), e2)) — upsample & concat folded in
    conv(stream, bt, 128, 1, 128 * S96, e2, 64, 64 * S192, IN(15), IN(16), d2a,
         192, 192, 192, 192, 3, 1, 1, 64, 1);
    // d1 = conv(concat(upsample2(d2), e1))
    conv(stream, d2a, 64, 1, 64 * S192, e1, 32, 32 * S384, IN(17), IN(18), d1a,
         384, 384, 384, 384, 3, 1, 1, 32, 1);
    // mask = sigmoid(conv(d1))
    conv(stream, d1a, 32, 0, 32 * S384, nullptr, 0, 0, IN(19), IN(20), maskb,
         384, 384, 384, 384, 3, 1, 1, 1, 2);

    // ---- attention branch ----
    conv(stream, maskb, 1, 0, S384, nullptr, 0, 0, IN(21), IN(22), atth,
         384, 384, 384, 384, 3, 1, 1, 16, 1);
    conv(stream, atth, 16, 0, 16 * S384, nullptr, 0, 0, IN(23), IN(24), attb,
         384, 384, 384, 384, 1, 1, 0, 1, 2);
    {
        long long tp = 16 * S384;
        mul_k<<<(unsigned)((tp + 255) / 256), 256, 0, stream>>>(maskb, attb, refb, tp);
    }

    // ---- border refine -> final_mask (written directly into d_out[64:]) ----
    conv(stream, refb, 1, 0, S384, nullptr, 0, 0, IN(25), IN(26), r1,
         384, 384, 384, 384, 3, 1, 1, 32, 1);
    conv(stream, r1, 32, 0, 32 * S384, nullptr, 0, 0, IN(27), IN(28), r2,
         384, 384, 384, 384, 3, 1, 1, 16, 1);
    conv(stream, r2, 16, 0, 16 * S384, nullptr, 0, 0, IN(29), IN(30), outp + 64,
         384, 384, 384, 384, 1, 1, 0, 1, 2);

    // ---- mask -> bbox, then BN-MLP head -> final_bbox (d_out[0:64]) ----
    bbox_k<<<16, 256, 0, stream>>>(outp + 64, mbbox);
    head_k<<<1, 256, 0, stream>>>(bboxf, mbbox,
                                  IN(31), IN(32), IN(33), IN(34),
                                  IN(35), IN(36), IN(37), IN(38),
                                  IN(39), IN(40), outp);
    #undef IN
}
